// ASISNativeAttention_81664508166398
// MI455X (gfx1250) — matrix operand order repaired and checked by reading, not compiled, not run on hardware
//
#include <hip/hip_runtime.h>
#include <hip/hip_bf16.h>

typedef __attribute__((ext_vector_type(16))) __bf16 v16bf;
typedef __attribute__((ext_vector_type(8)))  __bf16 v8bf;
typedef __attribute__((ext_vector_type(8)))  float  v8f;
typedef __attribute__((ext_vector_type(4))) unsigned int u32x4;
typedef __attribute__((ext_vector_type(8))) int          i32x8;
typedef __attribute__((ext_vector_type(4))) int          i32x4;

#define BSZ 2
#define SSZ 2048
#define DSZ 1024
#define HSZ 16
#define HDD 64
#define MSZ (BSZ*SSZ)   // 4096 rows for all big GEMMs

#if __has_builtin(__builtin_amdgcn_tensor_load_to_lds)
#define CDNA5_HAS_TDM 1
#else
#define CDNA5_HAS_TDM 0
#endif

union Frag16 { v16bf v; v8bf h[2]; };

__device__ __forceinline__ v8bf ld8(const __hip_bfloat16* p) {
    return *(const v8bf*)p;
}

#if CDNA5_HAS_TDM
// TDM: DMA a 64(n-rows) x 32(k-elems) bf16 tile, row stride 1024 elems, into LDS.
// D# built per CDNA5 ISA ch.8: group0 = {count/flags, lds_addr, global_addr, type},
// group1 = {data_size, tensor dims, tile dims, dim0 stride}. Groups 2/3 zero (2D).
// This toolchain exposes the 6-arg builtin (4 D# groups + trailing i32x8 + cpol).
__device__ __forceinline__ void tdm_load_b_tile(const __hip_bfloat16* gptr,
                                                unsigned lds_off) {
    unsigned long long ga = (unsigned long long)(uintptr_t)gptr;
    u32x4 g0;
    g0[0] = 1u;                                              // count=1 (valid), user mode
    g0[1] = lds_off;                                         // lds_addr (bytes)
    g0[2] = (unsigned)(ga & 0xFFFFFFFFu);                    // global_addr[31:0]
    g0[3] = (unsigned)((ga >> 32) & 0x01FFFFFFu) | (2u << 30); // addr[56:32] | type=2
    i32x8 g1;
    g1[0] = (int)(1u << 16);        // workgroup_mask=0, data_size=1 (2 bytes)
    g1[1] = (int)(1024u << 16);     // tensor_dim0[15:0]=1024  (bits 63:48)
    g1[2] = (int)(1024u << 16);     // tensor_dim0 hi=0 | tensor_dim1[15:0]=1024
    g1[3] = (int)(32u << 16);       // tensor_dim1 hi=0 | tile_dim0=32
    g1[4] = 64;                     // tile_dim1=64, tile_dim2=0
    g1[5] = 1024;                   // tensor_dim0_stride lo32 = 1024 elems
    g1[6] = 0;                      // stride hi16 | tensor_dim1_stride lo16 = 0
    g1[7] = 0;
    i32x4 gz4 = {0, 0, 0, 0};
    i32x8 gz8 = {0, 0, 0, 0, 0, 0, 0, 0};
    __builtin_amdgcn_tensor_load_to_lds(g0, g1, gz4, gz4, gz8, 0);
}
#endif

// ---------------------------------------------------------------- converts
__global__ void cvt_f32_bf16(const float* __restrict__ in,
                             __hip_bfloat16* __restrict__ out, int n) {
    int i = (blockIdx.x * blockDim.x + threadIdx.x) * 4;
    if (i + 3 < n) {
        float4 f = *(const float4*)(in + i);
        out[i + 0] = __float2bfloat16(f.x);
        out[i + 1] = __float2bfloat16(f.y);
        out[i + 2] = __float2bfloat16(f.z);
        out[i + 3] = __float2bfloat16(f.w);
    }
}

// W [K=D, N=D] row-major fp32  ->  Wt [N, K] row-major bf16
__global__ void cvt_w_transpose(const float* __restrict__ w,
                                __hip_bfloat16* __restrict__ wt) {
    int idx = blockIdx.x * blockDim.x + threadIdx.x;   // write-coalesced
    int n = idx >> 10, k = idx & (DSZ - 1);
    wt[idx] = __float2bfloat16(w[(size_t)k * DSZ + n]);
}

// ---------------------------------------------------------------- gates
__global__ void __launch_bounds__(256) gates_kernel(
        const float* __restrict__ x,
        const float* __restrict__ We, const float* __restrict__ be,
        const float* __restrict__ Ws, const float* __restrict__ bs,
        float* __restrict__ gate) {
    int b = blockIdx.x;
    __shared__ float xm[DSZ];
    __shared__ float red[2][256];
    int tid = threadIdx.x;
    for (int d = tid; d < DSZ; d += 256) {
        float s = 0.f;
        const float* xp = x + (size_t)b * SSZ * DSZ + d;
        for (int r = 0; r < SSZ; ++r) s += xp[(size_t)r * DSZ];
        xm[d] = s * (1.0f / (float)SSZ);
    }
    __syncthreads();
    int h = tid >> 4, c = tid & 15;
    float pe = 0.f, ps = 0.f;
    for (int d = c * 64; d < c * 64 + 64; ++d) {
        float v = xm[d];
        pe += v * We[(size_t)d * HSZ + h];
        ps += v * Ws[(size_t)d * HSZ + h];
    }
    red[0][tid] = pe; red[1][tid] = ps;
    __syncthreads();
    if (c == 0) {
        float e = 0.f, s = 0.f;
        for (int i = 0; i < 16; ++i) { e += red[0][(h << 4) + i]; s += red[1][(h << 4) + i]; }
        e += be[h]; s += bs[h];
        float ge = 1.f / (1.f + __expf(-e));
        float gs = 1.f / (1.f + __expf(-s));
        gate[b * HSZ + h] = ge * gs;
    }
}

// ---------------------------------------------------------------- WMMA GEMM
// out[m,n] = sum_k A[m,k]*Bt[n,k] + bias[n]      (M=4096, N=K=1024)
// mode 0: fp32 row-major [M,N];  mode 1: bf16 [B,H,S,HD];  mode 2: bf16 [B,H,HD,S]
__global__ void __launch_bounds__(256) gemm_bf16(
        const __hip_bfloat16* __restrict__ A,
        const __hip_bfloat16* __restrict__ Bt,
        const float* __restrict__ bias,
        float* __restrict__ outF,
        __hip_bfloat16* __restrict__ outB,
        int mode) {
    const int K = DSZ;
    __shared__ __align__(16) __hip_bfloat16 Bs[2][64 * 32];   // double-buffered B tile
    int wave = threadIdx.x >> 5, lane = threadIdx.x & 31;
    int tid = threadIdx.x;
    int nl = lane & 15, g = lane >> 4;
    int row0 = blockIdx.y * 128 + wave * 16;
    int col0 = blockIdx.x * 64;
    const __hip_bfloat16* Ar = A  + (size_t)(row0 + nl) * K;
    const __hip_bfloat16* Bg = Bt + (size_t)col0 * K;

#if CDNA5_HAS_TDM
    unsigned lds0 = (unsigned)(uintptr_t)&Bs[0][0];
    unsigned lds1 = (unsigned)(uintptr_t)&Bs[1][0];
    if (wave == 0) {                       // prefetch two k-steps ahead via TDM
        tdm_load_b_tile(Bg, lds0);
        tdm_load_b_tile(Bg + 32, lds1);
    }
#endif
    v8f z = {0.f,0.f,0.f,0.f,0.f,0.f,0.f,0.f};
    v8f acc[4] = {z, z, z, z};
    Frag16 a_cur, a_nxt;
    a_cur.h[0] = ld8(Ar + g * 8);
    a_cur.h[1] = ld8(Ar + 16 + g * 8);

    for (int kb = 0, it = 0; kb < K; kb += 32, ++it) {
        __hip_bfloat16* Bsc = &Bs[it & 1][0];
#if CDNA5_HAS_TDM
        if (wave == 0) __builtin_amdgcn_s_wait_tensorcnt((short)1);
        __syncthreads();                   // publish TDM-filled tile to all waves
#else
        {   // fallback: cooperative copy, one b128 per thread (256*16B = 4KB tile)
            int r = tid >> 2, sub = (tid & 3) * 8;
            *(v8bf*)(Bsc + r * 32 + sub) = ld8(Bg + (size_t)r * K + kb + sub);
            __syncthreads();
        }
#endif
        if (kb + 32 < K) {                 // register-pipelined A fragment
            a_nxt.h[0] = ld8(Ar + kb + 32 + g * 8);
            a_nxt.h[1] = ld8(Ar + kb + 48 + g * 8);
        }
        Frag16 bfr[4];
#pragma unroll
        for (int t = 0; t < 4; ++t) {
            const __hip_bfloat16* br = Bsc + (t * 16 + nl) * 32 + g * 8;
            bfr[t].h[0] = *(const v8bf*)br;
            bfr[t].h[1] = *(const v8bf*)(br + 16);
        }
#pragma unroll
        for (int t = 0; t < 4; ++t)
            acc[t] = __builtin_amdgcn_wmma_f32_16x16x32_bf16(
                         false, a_cur.v, false, bfr[t].v, (short)0, acc[t], false, false);
        __syncthreads();                   // everyone done reading Bs[it&1]
#if CDNA5_HAS_TDM
        if (wave == 0 && kb + 64 < K)
            tdm_load_b_tile(Bg + kb + 64, (it & 1) ? lds1 : lds0);
#endif
        if (kb + 32 < K) a_cur = a_nxt;
    }

#pragma unroll
    for (int t = 0; t < 4; ++t) {
        int col = col0 + t * 16 + nl;
        float bv = bias[col];
#pragma unroll
        for (int e = 0; e < 8; ++e) {
            int row = row0 + e + 8 * g;
            float val = acc[t][e] + bv;
            if (mode == 0) {
                outF[(size_t)row * DSZ + col] = val;
            } else {
                int b = row >> 11, s = row & (SSZ - 1);
                int h = col >> 6,  hd = col & (HDD - 1);
                size_t off = (mode == 1)
                    ? ((size_t)((b * HSZ + h) * SSZ + s) * HDD + hd)
                    : ((size_t)((b * HSZ + h) * HDD + hd) * SSZ + s);
                outB[off] = __float2bfloat16(val);
            }
        }
    }
}

// ---------------------------------------------------------------- flash attention
__device__ __forceinline__ void load_k_frags(const __hip_bfloat16* Kp, int kv,
                                             int nl, int g, Frag16 kf[2][2]) {
#pragma unroll
    for (int nt = 0; nt < 2; ++nt)
#pragma unroll
        for (int dh = 0; dh < 2; ++dh) {
            const __hip_bfloat16* kr =
                Kp + (size_t)(kv + nt * 16 + nl) * HDD + dh * 32 + g * 8;
            kf[nt][dh].h[0] = ld8(kr);
            kf[nt][dh].h[1] = ld8(kr + 16);
        }
}

// Q,K bf16 [B,H,S,64]; Vt bf16 [B,H,64,S]; ctx bf16 [B,S,D]
__global__ void __launch_bounds__(256) attn_kernel(
        const __hip_bfloat16* __restrict__ Q,
        const __hip_bfloat16* __restrict__ Kx,
        const __hip_bfloat16* __restrict__ Vt,
        const float* __restrict__ gate,
        __hip_bfloat16* __restrict__ ctx) {
    int bh = blockIdx.y;
    int b = bh >> 4, h = bh & 15;
    int wave = threadIdx.x >> 5, lane = threadIdx.x & 31;
    int nl = lane & 15, g = lane >> 4;
    int q0 = blockIdx.x * 128 + wave * 16;

    const __hip_bfloat16* Qp = Q  + (size_t)bh * SSZ * HDD;
    const __hip_bfloat16* Kp = Kx + (size_t)bh * SSZ * HDD;
    const __hip_bfloat16* Vp = Vt + (size_t)bh * HDD * SSZ;

    __shared__ __align__(16) __hip_bfloat16 Pb[8][16 * 32];
    __hip_bfloat16* myP = Pb[wave];

    Frag16 qa[2];
#pragma unroll
    for (int dh = 0; dh < 2; ++dh) {
        const __hip_bfloat16* qr = Qp + (size_t)(q0 + nl) * HDD + dh * 32 + g * 8;
        qa[dh].h[0] = ld8(qr);
        qa[dh].h[1] = ld8(qr + 16);
    }

    v8f z = {0.f,0.f,0.f,0.f,0.f,0.f,0.f,0.f};
    v8f o[4] = {z, z, z, z};
    float mrun[8], lrun[8];
#pragma unroll
    for (int e = 0; e < 8; ++e) { mrun[e] = -3.0e38f; lrun[e] = 0.f; }

    Frag16 kf[2][2], kn[2][2];
    load_k_frags(Kp, 0, nl, g, kf);        // software pipeline: K one step ahead

    for (int kv = 0; kv < SSZ; kv += 32) {
        // V loads issued early; consumed only after softmax VALU block
        Frag16 vf[4];
#pragma unroll
        for (int t = 0; t < 4; ++t) {
            const __hip_bfloat16* vr = Vp + (size_t)(t * 16 + nl) * SSZ + kv + g * 16;
            vf[t].h[0] = ld8(vr);
            vf[t].h[1] = ld8(vr + 8);
        }
        // scores: 16 q-rows x 32 kv-cols
        v8f s[2] = {z, z};
#pragma unroll
        for (int nt = 0; nt < 2; ++nt)
#pragma unroll
            for (int dh = 0; dh < 2; ++dh)
                s[nt] = __builtin_amdgcn_wmma_f32_16x16x32_bf16(
                            false, qa[dh].v, false, kf[nt][dh].v, (short)0, s[nt],
                            false, false);
        // prefetch next-step K fragments while softmax runs
        if (kv + 32 < SSZ) load_k_frags(Kp, kv + 32, nl, g, kn);

        // online softmax (scale 1/sqrt(64) = 0.125)
        float alpha[8];
#pragma unroll
        for (int e = 0; e < 8; ++e) {
            float s0 = s[0][e] * 0.125f, s1 = s[1][e] * 0.125f;
            float loc = fmaxf(s0, s1);
            for (int msk = 8; msk >= 1; msk >>= 1)
                loc = fmaxf(loc, __shfl_xor(loc, msk, 32));
            float mnew = fmaxf(mrun[e], loc);
            float a  = __expf(mrun[e] - mnew);
            float p0 = __expf(s0 - mnew);
            float p1 = __expf(s1 - mnew);
            float rs = p0 + p1;
            for (int msk = 8; msk >= 1; msk >>= 1)
                rs += __shfl_xor(rs, msk, 32);
            lrun[e] = lrun[e] * a + rs;
            mrun[e] = mnew;
            alpha[e] = a;
            int rm = e + 8 * g;
            myP[rm * 32 + nl]      = __float2bfloat16(p0);
            myP[rm * 32 + 16 + nl] = __float2bfloat16(p1);
        }
#pragma unroll
        for (int t = 0; t < 4; ++t)
#pragma unroll
            for (int e = 0; e < 8; ++e) o[t][e] *= alpha[e];
        // reload probs in A-layout (wave-coherent LDS round trip, DS in-order)
        Frag16 pa;
        pa.h[0] = *(const v8bf*)(myP + nl * 32 + g * 16);
        pa.h[1] = *(const v8bf*)(myP + nl * 32 + g * 16 + 8);
#pragma unroll
        for (int t = 0; t < 4; ++t)
            o[t] = __builtin_amdgcn_wmma_f32_16x16x32_bf16(
                       false, pa.v, false, vf[t].v, (short)0, o[t], false, false);
        if (kv + 32 < SSZ) {
#pragma unroll
            for (int nt = 0; nt < 2; ++nt)
#pragma unroll
                for (int dh = 0; dh < 2; ++dh) kf[nt][dh] = kn[nt][dh];
        }
    }
    float gt = gate[bh];
    float sc[8];
#pragma unroll
    for (int e = 0; e < 8; ++e) sc[e] = gt / lrun[e];
#pragma unroll
    for (int t = 0; t < 4; ++t) {
        int col = h * HDD + t * 16 + nl;
#pragma unroll
        for (int e = 0; e < 8; ++e) {
            int row = q0 + e + 8 * g;
            ctx[(size_t)(b * SSZ + row) * DSZ + col] = __float2bfloat16(o[t][e] * sc[e]);
        }
    }
}

// ---------------------------------------------------------------- launcher
extern "C" void kernel_launch(void* const* d_in, const int* in_sizes, int n_in,
                              void* d_out, int out_size, void* d_ws, size_t ws_size,
                              hipStream_t stream) {
    const float* x  = (const float*)d_in[0];
    const float* Wq = (const float*)d_in[1];  const float* bq = (const float*)d_in[2];
    const float* Wk = (const float*)d_in[3];  const float* bk = (const float*)d_in[4];
    const float* Wv = (const float*)d_in[5];  const float* bv = (const float*)d_in[6];
    const float* We = (const float*)d_in[7];  const float* be = (const float*)d_in[8];
    const float* Ws = (const float*)d_in[9];  const float* bs = (const float*)d_in[10];
    const float* Wo = (const float*)d_in[11]; const float* bo = (const float*)d_in[12];
    float* out = (float*)d_out;

    char* wsp = (char*)d_ws;
    size_t off = 0;
    auto alloc = [&](size_t bytes) -> char* {
        char* p = wsp + off;
        off += (bytes + 255) & ~(size_t)255;
        return p;
    };
    const size_t XE = (size_t)MSZ * DSZ;
    const size_t WE = (size_t)DSZ * DSZ;
    __hip_bfloat16* xb   = (__hip_bfloat16*)alloc(XE * 2);
    __hip_bfloat16* wqt  = (__hip_bfloat16*)alloc(WE * 2);
    __hip_bfloat16* wkt  = (__hip_bfloat16*)alloc(WE * 2);
    __hip_bfloat16* wvt  = (__hip_bfloat16*)alloc(WE * 2);
    __hip_bfloat16* wot  = (__hip_bfloat16*)alloc(WE * 2);
    __hip_bfloat16* qb   = (__hip_bfloat16*)alloc(XE * 2);
    __hip_bfloat16* kb   = (__hip_bfloat16*)alloc(XE * 2);
    __hip_bfloat16* vtb  = (__hip_bfloat16*)alloc(XE * 2);
    __hip_bfloat16* ctxb = (__hip_bfloat16*)alloc(XE * 2);
    float*          gtw  = (float*)alloc(BSZ * HSZ * sizeof(float));

    dim3 blk(256);
    cvt_f32_bf16<<<dim3((unsigned)(XE / 1024)), blk, 0, stream>>>(x, xb, (int)XE);
    cvt_w_transpose<<<dim3((unsigned)(WE / 256)), blk, 0, stream>>>(Wq, wqt);
    cvt_w_transpose<<<dim3((unsigned)(WE / 256)), blk, 0, stream>>>(Wk, wkt);
    cvt_w_transpose<<<dim3((unsigned)(WE / 256)), blk, 0, stream>>>(Wv, wvt);
    cvt_w_transpose<<<dim3((unsigned)(WE / 256)), blk, 0, stream>>>(Wo, wot);
    gates_kernel<<<dim3(BSZ), blk, 0, stream>>>(x, We, be, Ws, bs, gtw);
    dim3 ggrid(DSZ / 64, MSZ / 128);
    gemm_bf16<<<ggrid, blk, 0, stream>>>(xb, wqt, bq, nullptr, qb, 1);
    gemm_bf16<<<ggrid, blk, 0, stream>>>(xb, wkt, bk, nullptr, kb, 1);
    gemm_bf16<<<ggrid, blk, 0, stream>>>(xb, wvt, bv, nullptr, vtb, 2);
    dim3 agrid(SSZ / 128, BSZ * HSZ);
    attn_kernel<<<agrid, blk, 0, stream>>>(qb, kb, vtb, gtw, ctxb);
    gemm_bf16<<<ggrid, blk, 0, stream>>>(ctxb, wot, bo, out, nullptr, 0);
}
